// GNNModel_41832981463504
// MI455X (gfx1250) — compile-verified, matrix operand order checked by reading, not run on hardware
//
#include <hip/hip_runtime.h>
#include <hip/hip_fp16.h>
#include <math.h>

typedef __attribute__((ext_vector_type(16))) _Float16 v16h;
typedef __attribute__((ext_vector_type(8)))  float    v8f;

#define BZ    64
#define SEQ   512
#define NBR   16
#define DIM   256
#define NCLS  20
#define CHUNKS 16
#define SPB   (SEQ / CHUNKS)   // 32 seq positions per block

// ---------------------------------------------------------------------------
// Kernel 1: gather neighbor embeddings, max-combine, weighted mix, partial
// sum-pool over a chunk of the sequence.  L2-bandwidth bound -> b128 gathers.
// Grid: (BZ, CHUNKS); Block: (64, 4) = 256 threads = 8 waves.
// ---------------------------------------------------------------------------
__global__ __launch_bounds__(256)
void gnn_gather_pool(const int* __restrict__ X,
                     const int* __restrict__ NX,
                     const int* __restrict__ EW,
                     const float* __restrict__ node_emb,
                     const float* __restrict__ edge_w,
                     const float* __restrict__ node_w,
                     float* __restrict__ partial) {
  const int b     = blockIdx.x;
  const int chunk = blockIdx.y;
  const int tx    = threadIdx.x;   // 0..63 : float4 channel group
  const int ty    = threadIdx.y;   // 0..3  : seq sub-slot

  __shared__ int    lds_nb[4][NBR];
  __shared__ float  lds_ew[4][NBR];
  __shared__ float4 red[4][64];

  float ax = 0.f, ay = 0.f, az = 0.f, aw = 0.f;
  const int sbase = chunk * SPB;

  for (int s0 = 0; s0 < SPB; s0 += 4) {
    const int s = sbase + s0 + ty;
    const int i = b * SEQ + s;

    __syncthreads();               // protect LDS reuse across iterations
    if (tx < NBR) {
      const int nb = NX[i * NBR + tx];
      lds_nb[ty][tx] = nb;
      lds_ew[ty][tx] = edge_w[EW[i * NBR + tx]];
    }
    __syncthreads();

    const int   x  = X[i];
    const float nn = node_w[x];
    const float4 rn = *(const float4*)(node_emb + (size_t)x * DIM + 4 * tx);

    float mx = -INFINITY, my = -INFINITY, mz = -INFINITY, mw = -INFINITY;
#pragma unroll
    for (int j = 0; j < NBR; ++j) {
      if (j + 8 < NBR)   // gfx1250 global_prefetch_b8 on upcoming gather rows
        __builtin_prefetch(node_emb + (size_t)lds_nb[ty][j + 8] * DIM + 4 * tx, 0, 1);
      const int   nb = lds_nb[ty][j];
      const float ew = lds_ew[ty][j];
      const float4 ra = *(const float4*)(node_emb + (size_t)nb * DIM + 4 * tx);
      mx = fmaxf(mx, ra.x * ew);
      my = fmaxf(my, ra.y * ew);
      mz = fmaxf(mz, ra.z * ew);
      mw = fmaxf(mw, ra.w * ew);
    }
    const float c1 = 1.0f - nn;
    ax += c1 * mx + nn * rn.x;
    ay += c1 * my + nn * rn.y;
    az += c1 * mz + nn * rn.z;
    aw += c1 * mw + nn * rn.w;
  }

  red[ty][tx] = make_float4(ax, ay, az, aw);
  __syncthreads();
  if (ty == 0) {
    float4 r0 = red[0][tx], r1 = red[1][tx], r2 = red[2][tx], r3 = red[3][tx];
    float4 o;
    o.x = r0.x + r1.x + r2.x + r3.x;
    o.y = r0.y + r1.y + r2.y + r3.y;
    o.z = r0.z + r1.z + r2.z + r3.z;
    o.w = r0.w + r1.w + r2.w + r3.w;
    *(float4*)(partial + ((size_t)b * CHUNKS + chunk) * DIM + 4 * tx) = o;
  }
}

// ---------------------------------------------------------------------------
// Kernel 2: reduce chunk partials -> y[64][256]; FC (64x256)x(256x20) via
// v_wmma_f32_16x16x32_f16 (4 M-tiles x 2 N-tiles, one tile per wave, 8
// k-steps); bias + ReLU + log_softmax.  Single block, 256 threads = 8 waves.
// ---------------------------------------------------------------------------
__global__ __launch_bounds__(256)
void gnn_fc_logsoftmax(const float* __restrict__ partial,
                       const float* __restrict__ fc_W,
                       const float* __restrict__ fc_b,
                       float* __restrict__ out) {
  __shared__ _Float16 yh[BZ][DIM];     // 32 KB  (A matrix, f16)
  __shared__ _Float16 wh[32][DIM];     // 16 KB  (B matrix = fc_W, f16, N padded to 32)
  __shared__ float    logits[BZ][32];  //  8 KB

  const int tid = threadIdx.x;

  // Reduce 16 chunk partials -> y, convert to f16.
  for (int idx = tid; idx < BZ * (DIM / 4); idx += 256) {
    const int b  = idx / (DIM / 4);
    const int d4 = idx % (DIM / 4);
    float sx = 0.f, sy = 0.f, sz = 0.f, sw = 0.f;
    for (int c = 0; c < CHUNKS; ++c) {
      const float4 p = *(const float4*)(partial + ((size_t)b * CHUNKS + c) * DIM + 4 * d4);
      sx += p.x; sy += p.y; sz += p.z; sw += p.w;
    }
    yh[b][4 * d4 + 0] = (_Float16)sx;
    yh[b][4 * d4 + 1] = (_Float16)sy;
    yh[b][4 * d4 + 2] = (_Float16)sz;
    yh[b][4 * d4 + 3] = (_Float16)sw;
  }

  // Load fc_W (rows 20..31 zero-padded).
  for (int idx = tid; idx < 32 * (DIM / 4); idx += 256) {
    const int n  = idx / (DIM / 4);
    const int d4 = idx % (DIM / 4);
    float4 w = make_float4(0.f, 0.f, 0.f, 0.f);
    if (n < NCLS) w = *(const float4*)(fc_W + (size_t)n * DIM + 4 * d4);
    wh[n][4 * d4 + 0] = (_Float16)w.x;
    wh[n][4 * d4 + 1] = (_Float16)w.y;
    wh[n][4 * d4 + 2] = (_Float16)w.z;
    wh[n][4 * d4 + 3] = (_Float16)w.w;
  }
  __syncthreads();

  // WMMA: wave wv handles tile (m0, n0); lanes per the documented 16-bit
  // A-layout (lanes 0-15 = rows, hi half-wave carries K+8; B mirrored on N).
  const int wv   = tid >> 5;
  const int lane = tid & 31;
  const int row  = lane & 15;
  const int hi   = lane >> 4;
  const int m0   = (wv & 3) * 16;
  const int n0   = (wv >> 2) * 16;

  v8f c = {};
#pragma unroll
  for (int kk = 0; kk < DIM / 32; ++kk) {
    const int k0 = kk * 32;
    v16h a, bm;
#pragma unroll
    for (int e = 0; e < 16; ++e) {
      const int k = k0 + ((e >= 8) ? 16 : 0) + hi * 8 + (e & 7);
      a[e]  = yh[m0 + row][k];
      bm[e] = wh[n0 + row][k];
    }
    c = __builtin_amdgcn_wmma_f32_16x16x32_f16(
        /*neg_a=*/false, a, /*neg_b=*/false, bm,
        /*c_mod=*/(short)0, c, /*reuse_a=*/false, /*reuse_b=*/false);
  }

  // C/D layout: VGPR r -> M = r (lanes 0-15) / r+8 (lanes 16-31), N = lane%16.
#pragma unroll
  for (int r = 0; r < 8; ++r)
    logits[m0 + r + hi * 8][n0 + row] = c[r];
  __syncthreads();

  // Per-row bias + ReLU + log_softmax.
  if (tid < BZ) {
    float z[NCLS];
    float mx = -INFINITY;
#pragma unroll
    for (int n = 0; n < NCLS; ++n) {
      z[n] = fmaxf(logits[tid][n] + fc_b[n], 0.0f);
      mx = fmaxf(mx, z[n]);
    }
    float s = 0.f;
#pragma unroll
    for (int n = 0; n < NCLS; ++n) s += expf(z[n] - mx);
    const float lse = mx + logf(s);
#pragma unroll
    for (int n = 0; n < NCLS; ++n) out[tid * NCLS + n] = z[n] - lse;
  }
}

extern "C" void kernel_launch(void* const* d_in, const int* in_sizes, int n_in,
                              void* d_out, int out_size, void* d_ws, size_t ws_size,
                              hipStream_t stream) {
  const int*   X        = (const int*)d_in[0];
  const int*   NX       = (const int*)d_in[1];
  const int*   EW       = (const int*)d_in[2];
  const float* node_emb = (const float*)d_in[3];
  const float* edge_w   = (const float*)d_in[4];
  const float* node_w   = (const float*)d_in[5];
  const float* fc_W     = (const float*)d_in[6];
  const float* fc_b     = (const float*)d_in[7];
  float*       out      = (float*)d_out;

  float* partial = (float*)d_ws;   // BZ * CHUNKS * DIM floats = 1 MiB

  dim3 g1(BZ, CHUNKS), b1(64, 4);
  gnn_gather_pool<<<g1, b1, 0, stream>>>(X, NX, EW, node_emb, edge_w, node_w, partial);
  gnn_fc_logsoftmax<<<1, 256, 0, stream>>>(partial, fc_W, fc_b, out);
}
